// AttnEncoder_11141145166131
// MI455X (gfx1250) — matrix operand order repaired and checked by reading, not compiled, not run on hardware
//
#include <hip/hip_runtime.h>

// ---------------------------------------------------------------------------
// Types for CDNA5 WMMA (wave32): v_wmma_f32_16x16x32_bf16
// ---------------------------------------------------------------------------
typedef __bf16 bf16;
typedef bf16  v16bf __attribute__((ext_vector_type(16)));
typedef float v8f   __attribute__((ext_vector_type(8)));

#define WMMA_BF16(A, B, C) \
  __builtin_amdgcn_wmma_f32_16x16x32_bf16(false, (A), false, (B), (short)0, (C), false, false)

#define V8F_ZERO {0.f,0.f,0.f,0.f,0.f,0.f,0.f,0.f}

// Problem constants
static constexpr int Bq = 16, Lq = 64, Nq = 64, Hq = 512, Aq = 512, Vq = 1000, VPAD = 1024;
static constexpr int Mrows = Lq * Nq * Bq;       // 65536 rows of Fseq
static constexpr int MT = Mrows / 16;            // 4096 M-tiles

// Workspace layout (bytes, all 32B aligned)
static constexpr size_t OFF_FB  = 0;                                          // Fseq bf16 A-frags  64 MiB
static constexpr size_t OFF_P   = OFF_FB  + (size_t)Mrows * Aq * 2;           // P fp32            128 MiB
static constexpr size_t OFF_WAB = OFF_P   + (size_t)Mrows * Hq * 4;           // Wa  bf16 B-frags
static constexpr size_t OFF_WHB = OFF_WAB + (size_t)Aq * Hq * 2;              // Wh  bf16 B-frags
static constexpr size_t OFF_WPB = OFF_WHB + (size_t)Hq * Hq * 2;              // [Wih|Whh]^T bf16
static constexpr size_t OFF_LTB = OFF_WPB + (size_t)1024 * 2048 * 2;          // lt_W^T bf16 (padded)
static constexpr size_t OFF_YSB = OFF_LTB + (size_t)Hq * VPAD * 2;            // ys bf16 A-frags
static constexpr size_t OFF_LG  = OFF_YSB + (size_t)Lq * Bq * Hq * 2;         // logits fp32 padded
static constexpr size_t OFF_Q   = OFF_LG  + (size_t)Lq * Bq * VPAD * 4;
static constexpr size_t OFF_S   = OFF_Q   + (size_t)Bq * Hq * 4;
static constexpr size_t OFF_G   = OFF_S   + (size_t)Bq * Hq * 4;
static constexpr size_t OFF_H   = OFF_G   + (size_t)Bq * 4 * Hq * 4;
static constexpr size_t OFF_C   = OFF_H   + (size_t)2 * Bq * Hq * 4;

// Output layout (floats)
static constexpr size_t OUT_LOGITS = 0;
static constexpr size_t OUT_PROBS  = (size_t)Bq * Lq * Vq;                    // 1,024,000
static constexpr size_t OUT_HX     = OUT_PROBS + (size_t)Bq * Lq * Vq;        // 2,048,000
static constexpr size_t OUT_CX     = OUT_HX + (size_t)2 * Bq * Hq;            // 2,064,384
static constexpr size_t OUT_BETAS  = OUT_CX + (size_t)2 * Bq * Hq;            // 2,080,768

// ---------------------------------------------------------------------------
// ISA 7.12.2 16-bit A-matrix (16x32) per-lane layout helpers.
// lane l: m = l&15, g = l>>4; slot j (0..15, two bf16 per VGPR):
//   K = (v<4) ? 2v+half+8g : 2(v-4)+half+16+8g   with v=j>>1, half=j&1
// ---------------------------------------------------------------------------
static __device__ __forceinline__ int a_k(int lane, int j) {
  int g = lane >> 4, v = j >> 1, h = j & 1;
  return (v < 4) ? (2 * v + h + 8 * g) : (2 * (v - 4) + h + 16 + 8 * g);
}
static __device__ __forceinline__ void a_inv(int m, int k, int& lane, int& j) {
  int g = (k >> 3) & 1;
  int kk = k - 8 * g;              // in {0..7, 16..23}
  j = (kk < 8) ? kk : (kk - 8);
  lane = g * 16 + m;
}
// B-matrix (32x16): lane l: n = l&15, K = a_k(l, j), same order as A

// ---------------------------------------------------------------------------
// Prep kernels
// ---------------------------------------------------------------------------
__global__ void k_init_hc(const float* __restrict__ h0h, const float* __restrict__ h0c,
                          float* __restrict__ h, float* __restrict__ c) {
  int idx = blockIdx.x * 256 + threadIdx.x;          // (layer, b, k)
  if (idx >= 2 * Bq * Hq) return;
  int k = idx & 511, b = (idx >> 9) & 15, layer = idx >> 13;
  h[idx] = h0h[((size_t)b * 2 + layer) * Hq + k];
  c[idx] = h0c[((size_t)b * 2 + layer) * Hq + k];
}

// Fs (B,L,F,8,8) -> bf16 A-fragment-major [mtile][kf=16][lane=32][16]
__global__ void k_pack_F(const float* __restrict__ Fs, bf16* __restrict__ FB) {
  int cchunk = blockIdx.x * 256 + threadIdx.x;
  if (cchunk >= MT * 16 * 32) return;
  int lane = cchunk & 31, kf = (cchunk >> 5) & 15, mt = cchunk >> 9;
  int m = lane & 15;
  int r = mt * 16 + m;                               // row = ((l*64 + n)*16 + b)
  int b = r & 15, n = (r >> 4) & 63, l = r >> 10;
  bf16* d = FB + (size_t)cchunk * 16;
#pragma unroll
  for (int j = 0; j < 16; ++j) {
    int a = kf * 32 + a_k(lane, j);
    d[j] = (bf16)Fs[(((size_t)(b * 64 + l) * 512 + a) << 6) + n];
  }
}

// W (512,512) row-major [k][n] -> bf16 B-fragment-major [kf=16][nt=32][lane][16]
__global__ void k_pack_B512(const float* __restrict__ W, bf16* __restrict__ dst) {
  int cchunk = blockIdx.x * 256 + threadIdx.x;
  if (cchunk >= 16 * 32 * 32) return;
  int lane = cchunk & 31, nt = (cchunk >> 5) & 31, kf = cchunk >> 10;
  int g = lane >> 4, nl = lane & 15;
  bf16* d = dst + (size_t)cchunk * 16;
#pragma unroll
  for (int j = 0; j < 16; ++j) {
    int kg = kf * 32 + a_k(lane, j);
    d[j] = (bf16)W[(size_t)kg * 512 + nt * 16 + nl];
  }
}

// [W_ih | W_hh]^T : K=1024, N=2048 -> [kf=32][nt=128][lane][16]
__global__ void k_pack_Wp(const float* __restrict__ Wih, const float* __restrict__ Whh,
                          bf16* __restrict__ dst) {
  int cchunk = blockIdx.x * 256 + threadIdx.x;
  if (cchunk >= 32 * 128 * 32) return;
  int lane = cchunk & 31, nt = (cchunk >> 5) & 127, kf = cchunk >> 12;
  int g = lane >> 4, ncol = nt * 16 + (lane & 15);
  bf16* d = dst + (size_t)cchunk * 16;
#pragma unroll
  for (int j = 0; j < 16; ++j) {
    int kg = kf * 32 + a_k(lane, j);
    float v = (kg < 512) ? Wih[(size_t)ncol * 512 + kg] : Whh[(size_t)ncol * 512 + (kg - 512)];
    d[j] = (bf16)v;
  }
}

// lt_W^T : K=512, N=1024 (cols >=1000 zero) -> [kf=16][nt=64][lane][16]
__global__ void k_pack_ltW(const float* __restrict__ ltW, bf16* __restrict__ dst) {
  int cchunk = blockIdx.x * 256 + threadIdx.x;
  if (cchunk >= 16 * 64 * 32) return;
  int lane = cchunk & 31, nt = (cchunk >> 5) & 63, kf = cchunk >> 11;
  int g = lane >> 4, ncol = nt * 16 + (lane & 15);
  bf16* d = dst + (size_t)cchunk * 16;
#pragma unroll
  for (int j = 0; j < 16; ++j) {
    int kg = kf * 32 + a_k(lane, j);
    d[j] = (ncol < Vq) ? (bf16)ltW[(size_t)ncol * 512 + kg] : (bf16)0.f;
  }
}

// ---------------------------------------------------------------------------
// Big GEMM: P = Fseq @ Wa   (65536 x 512 x 512), one 16x16 C-tile row x 8 cols
// per wave. __launch_bounds__(256,1): keep 8 fp32 accumulators + pipelined
// B-fragments register-resident (no scratch spill) — single block per WGP is
// fine, grid is 2048 blocks and WMMA latency is hidden by the 8 waves.
// ---------------------------------------------------------------------------
__global__ void __launch_bounds__(256, 1)
k_gemm_P(const bf16* __restrict__ FB, const bf16* __restrict__ WaB,
         float* __restrict__ P) {
  int wave = (blockIdx.x * 256 + threadIdx.x) >> 5;
  int lane = threadIdx.x & 31;
  int mt = wave >> 2;            // 4096 M-tiles
  int ng = wave & 3;             // N-tile group of 8 (covers 512 cols)
  v8f acc[8] = {V8F_ZERO, V8F_ZERO, V8F_ZERO, V8F_ZERO,
                V8F_ZERO, V8F_ZERO, V8F_ZERO, V8F_ZERO};
  for (int kf = 0; kf < 16; ++kf) {
    const bf16* ap = FB + ((((size_t)mt * 16 + kf) * 32 + lane) << 4);
    v16bf a = *(const v16bf*)ap;
    if (kf < 15) __builtin_prefetch(ap + 512, 0, 1);   // next kf A-chunk (global_prefetch)
#pragma unroll
    for (int t = 0; t < 8; ++t) {
      v16bf b = *(const v16bf*)(WaB + ((((size_t)kf * 32 + ng * 8 + t) * 32 + lane) << 4));
      acc[t] = WMMA_BF16(a, b, acc[t]);
    }
  }
  int mrow = (lane >> 4) * 8, ncol = lane & 15;
#pragma unroll
  for (int t = 0; t < 8; ++t)
#pragma unroll
    for (int r = 0; r < 8; ++r)
      P[(size_t)(mt * 16 + mrow + r) * 512 + (ng * 8 + t) * 16 + ncol] = acc[t][r];
}

// ---------------------------------------------------------------------------
// q = h_top @ Wh   (16 x 512 x 512); A packed from fp32 h via LDS; 4 blocks x 8 waves
// ---------------------------------------------------------------------------
__global__ void __launch_bounds__(256, 1)
k_q(const float* __restrict__ h1, const bf16* __restrict__ WhB,
    float* __restrict__ q) {
  __shared__ __align__(32) bf16 As[16 * 512];
  int tid = threadIdx.x;
  for (int idx = tid; idx < 8192; idx += 256) {
    int j = idx & 15, lane = (idx >> 4) & 31, kf = idx >> 9;
    As[idx] = (bf16)h1[(size_t)(lane & 15) * 512 + kf * 32 + a_k(lane, j)];
  }
  __syncthreads();
  int wave = tid >> 5, lane = tid & 31;
  int nt = blockIdx.x * 8 + wave;
  v8f acc = V8F_ZERO;
  for (int kf = 0; kf < 16; ++kf) {
    v16bf a = *(const v16bf*)(&As[(kf * 32 + lane) << 4]);
    v16bf b = *(const v16bf*)(WhB + ((((size_t)kf * 32 + nt) * 32 + lane) << 4));
    acc = WMMA_BF16(a, b, acc);
  }
  int mrow = (lane >> 4) * 8, ncol = lane & 15;
#pragma unroll
  for (int r = 0; r < 8; ++r)
    q[(size_t)(mrow + r) * 512 + nt * 16 + ncol] = acc[r];
}

// ---------------------------------------------------------------------------
// score -> softmax(N) -> mask -> pooled s ; one block per batch b
// ---------------------------------------------------------------------------
__global__ void k_score_s(const float* __restrict__ P, const float* __restrict__ q,
                          const float* __restrict__ vv, const float* __restrict__ Ms,
                          const float* __restrict__ Fs, float* __restrict__ s,
                          float* __restrict__ betas, int t) {
  int b = blockIdx.x;
  int tid = threadIdx.x, wave = tid >> 5, lane = tid & 31;
  __shared__ float sc[64], w[64], wsum_s;
  for (int qq = 0; qq < 8; ++qq) {
    int n = wave * 8 + qq;
    size_t rowP = ((size_t)(t * 64 + n) * 16 + b) * 512;
    float p = 0.f;
    for (int h = lane; h < 512; h += 32)
      p += vv[h] * tanhf(P[rowP + h] + q[(size_t)b * 512 + h]);
#pragma unroll
    for (int off = 16; off > 0; off >>= 1) p += __shfl_xor(p, off, 32);
    if (lane == 0) sc[n] = p;
  }
  __syncthreads();
  if (tid == 0) {
    float mx = sc[0];
    for (int n = 1; n < 64; ++n) mx = fmaxf(mx, sc[n]);
    float sum = 0.f;
    for (int n = 0; n < 64; ++n) { float e = expf(sc[n] - mx); w[n] = e; sum += e; }
    float ws = 0.f;
    for (int n = 0; n < 64; ++n) {
      w[n] = (w[n] / sum) * Ms[((size_t)b * 64 + t) * 64 + n];   // GAMMA == 1
      ws += w[n];
    }
    wsum_s = fmaxf(ws, 1e-5f);
  }
  __syncthreads();
  if (tid < 64) betas[((size_t)b * 64 + t) * 64 + tid] = w[tid];
  for (int a = tid; a < 512; a += 256) {
    size_t base = ((size_t)(b * 64 + t) * 512 + a) << 6;         // Fs[(b,l,a,:)] contiguous in n
    float acc = 0.f;
    for (int n = 0; n < 64; ++n) acc += w[n] * Fs[base + n];
    s[(size_t)b * 512 + a] = acc / wsum_s;
  }
}

// ---------------------------------------------------------------------------
// LSTM gate GEMM: g = [x|h] @ [W_ih|W_hh]^T  (16 x 1024 x 2048); 8 blocks x 8 waves
// ---------------------------------------------------------------------------
__global__ void __launch_bounds__(256, 1)
k_lstm_gemm(const float* __restrict__ in, const float* __restrict__ hid,
            const bf16* __restrict__ WpB, float* __restrict__ g) {
  __shared__ __align__(32) bf16 As[16 * 1024];
  int tid = threadIdx.x;
  for (int idx = tid; idx < 16384; idx += 256) {
    int j = idx & 15, lane = (idx >> 4) & 31, kf = idx >> 9;
    int m = lane & 15;
    int kg = kf * 32 + a_k(lane, j);
    float v = (kg < 512) ? in[(size_t)m * 512 + kg] : hid[(size_t)m * 512 + (kg - 512)];
    As[idx] = (bf16)v;
  }
  __syncthreads();
  int wave = tid >> 5, lane = tid & 31;
  int nt0 = blockIdx.x * 16 + wave * 2;
  v8f acc0 = V8F_ZERO, acc1 = V8F_ZERO;
  for (int kf = 0; kf < 32; ++kf) {
    v16bf a  = *(const v16bf*)(&As[(kf * 32 + lane) << 4]);
    v16bf b0 = *(const v16bf*)(WpB + ((((size_t)kf * 128 + nt0)     * 32 + lane) << 4));
    v16bf b1 = *(const v16bf*)(WpB + ((((size_t)kf * 128 + nt0 + 1) * 32 + lane) << 4));
    acc0 = WMMA_BF16(a, b0, acc0);
    acc1 = WMMA_BF16(a, b1, acc1);
  }
  int mrow = (lane >> 4) * 8, ncol = lane & 15;
#pragma unroll
  for (int r = 0; r < 8; ++r) {
    g[(size_t)(mrow + r) * 2048 + nt0 * 16 + ncol]        = acc0[r];
    g[(size_t)(mrow + r) * 2048 + (nt0 + 1) * 16 + ncol]  = acc1[r];
  }
}

// gates + state update; for layer 1 also pack h into ys bf16 A-fragments (mtile = t)
__global__ void k_lstm_update(const float* __restrict__ g, const float* __restrict__ bih,
                              const float* __restrict__ bhh, float* __restrict__ h,
                              float* __restrict__ c, bf16* __restrict__ ysB, int t) {
  int idx = blockIdx.x * 256 + threadIdx.x;
  if (idx >= Bq * Hq) return;
  int b = idx >> 9, hh = idx & 511;
  const float* gr = g + (size_t)b * 2048;
  float iv = gr[hh]         + bih[hh]         + bhh[hh];
  float fv = gr[512 + hh]   + bih[512 + hh]   + bhh[512 + hh];
  float gv = gr[1024 + hh]  + bih[1024 + hh]  + bhh[1024 + hh];
  float ov = gr[1536 + hh]  + bih[1536 + hh]  + bhh[1536 + hh];
  float si = 1.f / (1.f + expf(-iv));
  float sf = 1.f / (1.f + expf(-fv));
  float so = 1.f / (1.f + expf(-ov));
  float cn = sf * c[idx] + si * tanhf(gv);
  float hn = so * tanhf(cn);
  c[idx] = cn;
  h[idx] = hn;
  if (ysB) {                       // row = t*16 + b of ys (L*B, H)
    int kf = hh >> 5, k = hh & 31, lane, j;
    a_inv(b, k, lane, j);
    ysB[((((size_t)t * 16 + kf) * 32) + lane) * 16 + j] = (bf16)hn;
  }
}

// ---------------------------------------------------------------------------
// logits = ys @ lt_W^T  (1024 x 512 x 1024-padded); 64 blocks x 8 waves
// ---------------------------------------------------------------------------
__global__ void __launch_bounds__(256, 1)
k_gemm_logits(const bf16* __restrict__ ysB, const bf16* __restrict__ ltWB,
              float* __restrict__ Lg) {
  int wave = (blockIdx.x * 256 + threadIdx.x) >> 5;
  int lane = threadIdx.x & 31;
  int mt = wave >> 3;            // 64 M-tiles
  int ng = wave & 7;             // 8 groups of 8 N-tiles -> 1024 cols
  v8f acc[8] = {V8F_ZERO, V8F_ZERO, V8F_ZERO, V8F_ZERO,
                V8F_ZERO, V8F_ZERO, V8F_ZERO, V8F_ZERO};
  for (int kf = 0; kf < 16; ++kf) {
    v16bf a = *(const v16bf*)(ysB + ((((size_t)mt * 16 + kf) * 32 + lane) << 4));
#pragma unroll
    for (int tN = 0; tN < 8; ++tN) {
      v16bf b = *(const v16bf*)(ltWB + ((((size_t)kf * 64 + ng * 8 + tN) * 32 + lane) << 4));
      acc[tN] = WMMA_BF16(a, b, acc[tN]);
    }
  }
  int mrow = (lane >> 4) * 8, ncol = lane & 15;
#pragma unroll
  for (int tN = 0; tN < 8; ++tN)
#pragma unroll
    for (int r = 0; r < 8; ++r)
      Lg[(size_t)(mt * 16 + mrow + r) * VPAD + (ng * 8 + tN) * 16 + ncol] = acc[tN][r];
}

// bias + softmax over V, transpose (L,B,V)->(B,L,V); one block per (l,b) row
__global__ void k_finalize(const float* __restrict__ Lg, const float* __restrict__ ltb,
                           float* __restrict__ outL, float* __restrict__ outP) {
  int row = blockIdx.x;                // l*16 + b
  int l = row >> 4, b = row & 15;
  int tid = threadIdx.x;
  __shared__ float red[256];
  const float* src = Lg + (size_t)row * VPAD;
  float mx = -3.4e38f;
  for (int v = tid; v < Vq; v += 256) mx = fmaxf(mx, src[v] + ltb[v]);
  red[tid] = mx; __syncthreads();
  for (int st = 128; st > 0; st >>= 1) {
    if (tid < st) red[tid] = fmaxf(red[tid], red[tid + st]);
    __syncthreads();
  }
  mx = red[0]; __syncthreads();
  float sum = 0.f;
  for (int v = tid; v < Vq; v += 256) sum += expf(src[v] + ltb[v] - mx);
  red[tid] = sum; __syncthreads();
  for (int st = 128; st > 0; st >>= 1) {
    if (tid < st) red[tid] += red[tid + st];
    __syncthreads();
  }
  sum = red[0];
  size_t obase = ((size_t)b * 64 + l) * Vq;
  for (int v = tid; v < Vq; v += 256) {
    float x = src[v] + ltb[v];
    outL[obase + v] = x;
    outP[obase + v] = expf(x - mx) / sum;
  }
}

__global__ void k_copy_hc(const float* __restrict__ h, const float* __restrict__ c,
                          float* __restrict__ outHx, float* __restrict__ outCx) {
  int idx = blockIdx.x * 256 + threadIdx.x;
  if (idx >= 2 * Bq * Hq) return;
  outHx[idx] = h[idx];
  outCx[idx] = c[idx];
}

// ---------------------------------------------------------------------------
// Launch
// ---------------------------------------------------------------------------
extern "C" void kernel_launch(void* const* d_in, const int* in_sizes, int n_in,
                              void* d_out, int out_size, void* d_ws, size_t ws_size,
                              hipStream_t stream) {
  (void)in_sizes; (void)n_in; (void)out_size; (void)ws_size;
  const float* Fs  = (const float*)d_in[0];
  const float* h0h = (const float*)d_in[1];
  const float* h0c = (const float*)d_in[2];
  const float* Ms  = (const float*)d_in[3];
  const float* Wa  = (const float*)d_in[4];
  const float* Wh  = (const float*)d_in[5];
  const float* vv  = (const float*)d_in[6];
  const float* Wih = (const float*)d_in[7];
  const float* Whh = (const float*)d_in[8];
  const float* bih = (const float*)d_in[9];
  const float* bhh = (const float*)d_in[10];
  const float* ltW = (const float*)d_in[11];
  const float* ltb = (const float*)d_in[12];
  float* out = (float*)d_out;

  char* ws = (char*)d_ws;
  bf16*  FB   = (bf16*)(ws + OFF_FB);
  float* P    = (float*)(ws + OFF_P);
  bf16*  WaB  = (bf16*)(ws + OFF_WAB);
  bf16*  WhB  = (bf16*)(ws + OFF_WHB);
  bf16*  WpB  = (bf16*)(ws + OFF_WPB);
  bf16*  ltWB = (bf16*)(ws + OFF_LTB);
  bf16*  ysB  = (bf16*)(ws + OFF_YSB);
  float* Lg   = (float*)(ws + OFF_LG);
  float* qb   = (float*)(ws + OFF_Q);
  float* sb   = (float*)(ws + OFF_S);
  float* gb   = (float*)(ws + OFF_G);
  float* hb   = (float*)(ws + OFF_H);
  float* cb   = (float*)(ws + OFF_C);

  // Prep: state init + bf16 fragment packing of all GEMM operands
  k_init_hc <<<64,   256, 0, stream>>>(h0h, h0c, hb, cb);
  k_pack_F  <<<8192, 256, 0, stream>>>(Fs, FB);
  k_pack_B512<<<64,  256, 0, stream>>>(Wa, WaB);
  k_pack_B512<<<64,  256, 0, stream>>>(Wh, WhB);
  k_pack_Wp <<<512,  256, 0, stream>>>(Wih, Whh, WpB);
  k_pack_ltW<<<128,  256, 0, stream>>>(ltW, ltWB);

  // Hoisted attention projection: P = Fseq @ Wa over all 64 steps at once
  k_gemm_P<<<2048, 256, 0, stream>>>(FB, WaB, P);

  // Sequential recurrence (64 steps)
  for (int t = 0; t < Lq; ++t) {
    k_q        <<<4,  256, 0, stream>>>(hb + 8192, WhB, qb);                      // q = h_top @ Wh
    k_score_s  <<<16, 256, 0, stream>>>(P, qb, vv, Ms, Fs, sb, out + OUT_BETAS, t);
    k_lstm_gemm<<<8,  256, 0, stream>>>(sb, hb,        WpB, gb);                  // layer 0
    k_lstm_update<<<32, 256, 0, stream>>>(gb, bih, bhh, hb,        cb,        nullptr, t);
    k_lstm_gemm<<<8,  256, 0, stream>>>(hb, hb + 8192, WpB, gb);                  // layer 1 (input = new h0)
    k_lstm_update<<<32, 256, 0, stream>>>(gb, bih, bhh, hb + 8192, cb + 8192, ysB, t);
  }

  // Output head
  k_gemm_logits<<<64,   256, 0, stream>>>(ysB, ltWB, Lg);
  k_finalize  <<<1024, 256, 0, stream>>>(Lg, ltb, out + OUT_LOGITS, out + OUT_PROBS);
  k_copy_hc   <<<64,   256, 0, stream>>>(hb, cb, out + OUT_HX, out + OUT_CX);
}
